// LPGNNLayer_51539608255
// MI455X (gfx1250) — matrix operand order repaired and checked by reading, not compiled, not run on hardware
//
#include <hip/hip_runtime.h>

// ---------------------------------------------------------------------------
// LPGNN layer for gfx1250 (MI455X).  GEMMs run on v_wmma_f32_16x16x32_f16,
// one wave32 per 32x32 output block (4 WMMAs per K-step, 2x fragment reuse).
// ---------------------------------------------------------------------------

#define N_NODES 8000
#define N_EDGES 32000
#define DMODEL  128
#define NHEAD   8
#define HD      16
#define HID     512
#define SCALE   0.25f      // 1/sqrt(16)
#define LN_EPS  1e-5f

typedef _Float16 v16h __attribute__((ext_vector_type(16)));
typedef _Float16 h8   __attribute__((ext_vector_type(8)));
typedef float    v8f  __attribute__((ext_vector_type(8)));

// ---- order-preserving float <-> uint encoding for atomicMax on floats ------
__device__ __forceinline__ unsigned f2ord(float f) {
    unsigned u = __float_as_uint(f);
    return (u & 0x80000000u) ? ~u : (u | 0x80000000u);
}
__device__ __forceinline__ float ord2f(unsigned u) {
    unsigned v = (u & 0x80000000u) ? (u & 0x7FFFFFFFu) : ~u;
    return __uint_as_float(v);
}

__device__ __forceinline__ v16h load_afrag(const _Float16* p) {
    // A fragment (16-bit A 16x32 layout): elements 0..7 -> K+0..7,
    // elements 8..15 -> K+16..23 (caller applies lane-dependent base).
    h8 a0 = *(const h8*)(p);
    h8 a1 = *(const h8*)(p + 16);
    v16h av;
#pragma unroll
    for (int i = 0; i < 8; ++i) { av[i] = a0[i]; av[i + 8] = a1[i]; }
    return av;
}

// ---------------------------------------------------------------------------
// GEMM:  C[M x N] = A_f16[M x K] * Wt_f16[N x K]^T + bias (+res) (relu?)
// wave32 -> 32x32 output block; K in steps of 32 via WMMA f16.
// ---------------------------------------------------------------------------
__global__ void gemm_wmma_f16(const _Float16* __restrict__ A,
                              const _Float16* __restrict__ Wt,
                              const float* __restrict__ bias,
                              const float* __restrict__ res,
                              float* __restrict__ Cf,
                              _Float16* __restrict__ Ch,
                              int M, int N, int K, int relu)
{
    const int lane   = threadIdx.x & 31;
    const int wid    = (int)((blockIdx.x * blockDim.x + threadIdx.x) >> 5);
    const int numNT2 = N >> 5;
    const int numMT2 = M >> 5;
    if (wid >= numMT2 * numNT2) return;
    const int mt = (wid / numNT2) << 5;   // row base of 32x32 block
    const int nt = (wid % numNT2) << 5;   // col base of 32x32 block

    const int r16 = lane & 15;
    const int hi  = lane >> 4;           // 0 or 1
    const int ka  = hi << 3;             // A: K-subblock base 0 / 8
    const int kb  = hi << 3;             // B: K-subblock base 0 / 8

    const _Float16* ap0 = A  + (size_t)(mt + r16)      * K + ka;
    const _Float16* ap1 = A  + (size_t)(mt + 16 + r16) * K + ka;
    const _Float16* bp0 = Wt + (size_t)(nt + r16)      * K + kb;
    const _Float16* bp1 = Wt + (size_t)(nt + 16 + r16) * K + kb;

    v8f acc00 = {}, acc01 = {}, acc10 = {}, acc11 = {};
    for (int k0 = 0; k0 < K; k0 += 32) {
        v16h av0 = load_afrag(ap0 + k0);
        v16h av1 = load_afrag(ap1 + k0);
        v16h bv0 = load_afrag(bp0 + k0);
        v16h bv1 = load_afrag(bp1 + k0);
        acc00 = __builtin_amdgcn_wmma_f32_16x16x32_f16(false, av0, false, bv0,
                                                       (short)0, acc00, false, false);
        acc01 = __builtin_amdgcn_wmma_f32_16x16x32_f16(false, av0, false, bv1,
                                                       (short)0, acc01, false, false);
        acc10 = __builtin_amdgcn_wmma_f32_16x16x32_f16(false, av1, false, bv0,
                                                       (short)0, acc10, false, false);
        acc11 = __builtin_amdgcn_wmma_f32_16x16x32_f16(false, av1, false, bv1,
                                                       (short)0, acc11, false, false);
    }

    // C/D layout: lane -> column (lane&15), VGPR r -> row r + (lane>=16 ? 8:0)
    v8f accs[2][2] = {{acc00, acc01}, {acc10, acc11}};
#pragma unroll
    for (int ti = 0; ti < 2; ++ti) {
#pragma unroll
        for (int tj = 0; tj < 2; ++tj) {
            const int col   = nt + tj * 16 + r16;
            const int mbase = mt + ti * 16 + (hi << 3);
            const float bc  = bias ? bias[col] : 0.0f;
#pragma unroll
            for (int r = 0; r < 8; ++r) {
                const size_t o = (size_t)(mbase + r) * N + col;
                float v = accs[ti][tj][r] + bc;
                if (res)  v += res[o];
                if (relu) v = v > 0.0f ? v : 0.0f;
                if (Cf) Cf[o] = v;
                if (Ch) Ch[o] = (_Float16)v;
            }
        }
    }
}

// ---------------------------------------------------------------------------
// LayerNorm over D=128, one wave32 per row, f16 output (GEMM input).
// ---------------------------------------------------------------------------
__global__ void ln_to_f16(const float* __restrict__ x,
                          const float* __restrict__ g,
                          const float* __restrict__ b,
                          _Float16* __restrict__ out, int n)
{
    const int wave = (int)((blockIdx.x * blockDim.x + threadIdx.x) >> 5);
    const int lane = threadIdx.x & 31;
    if (wave >= n) return;
    const float* row = x + (size_t)wave * DMODEL;
    float v[4], s = 0.0f, ss = 0.0f;
#pragma unroll
    for (int j = 0; j < 4; ++j) {
        v[j] = row[lane * 4 + j];
        s += v[j];
        ss += v[j] * v[j];
    }
#pragma unroll
    for (int m = 16; m > 0; m >>= 1) {
        s  += __shfl_xor(s,  m, 32);
        ss += __shfl_xor(ss, m, 32);
    }
    const float mean = s * (1.0f / DMODEL);
    const float var  = ss * (1.0f / DMODEL) - mean * mean;
    const float inv  = rsqrtf(var + LN_EPS);
    _Float16* orow = out + (size_t)wave * DMODEL;
#pragma unroll
    for (int j = 0; j < 4; ++j) {
        const int d = lane * 4 + j;
        orow[d] = (_Float16)((v[j] - mean) * inv * g[d] + b[d]);
    }
}

// weight f32[din][dout] -> f16 Wt[dout][din]
__global__ void w_to_f16t(const float* __restrict__ w, _Float16* __restrict__ wt,
                          int din, int dout)
{
    const int idx = (int)(blockIdx.x * blockDim.x + threadIdx.x);
    if (idx >= din * dout) return;
    const int i = idx / dout, o = idx % dout;
    wt[(size_t)o * din + i] = (_Float16)w[idx];
}

__global__ void f32_to_f16(const float* __restrict__ in, _Float16* __restrict__ out,
                           int n)
{
    const int t = (int)(blockIdx.x * blockDim.x + threadIdx.x);
    if (t < n) out[t] = (_Float16)in[t];
}

// out[i,:] += nodes[idx[i],:]
__global__ void gather_add(float* __restrict__ out, const float* __restrict__ nodes,
                           const int* __restrict__ idx, int n)
{
    const long long t = (long long)blockIdx.x * blockDim.x + threadIdx.x;
    if (t >= (long long)n * DMODEL) return;
    const int i = (int)(t >> 7), d = (int)(t & 127);
    out[t] += nodes[(size_t)idx[i] * DMODEL + d];
}

// per (i,head): s = SCALE * <Q[a], K[b]>_head ; segment atomic-max
__global__ void scores_k(const float* __restrict__ Q, const float* __restrict__ Kk,
                         const int* __restrict__ pairs, const int* __restrict__ seg,
                         float* __restrict__ s, unsigned* __restrict__ mEnc, int n)
{
    const long long t = (long long)blockIdx.x * blockDim.x + threadIdx.x;
    if (t >= (long long)n * NHEAD) return;
    const int i = (int)(t >> 3), h = (int)(t & 7);
    int a, b2, sg;
    if (pairs) { a = pairs[2 * i]; b2 = pairs[2 * i + 1]; sg = a; }
    else       { a = i; b2 = i; sg = seg[i]; }
    const float* q = Q  + (size_t)a  * DMODEL + h * HD;
    const float* k = Kk + (size_t)b2 * DMODEL + h * HD;
    float acc = 0.0f;
#pragma unroll
    for (int d = 0; d < HD; ++d) acc += q[d] * k[d];
    acc *= SCALE;
    s[t] = acc;
    atomicMax(&mEnc[(size_t)sg * NHEAD + h], f2ord(acc));
}

// e = exp(s - m[seg]); s <- e; z[seg] += e
__global__ void expsum_k(float* __restrict__ s, const unsigned* __restrict__ mEnc,
                         float* __restrict__ z, const int* __restrict__ pairs,
                         const int* __restrict__ seg, int n)
{
    const long long t = (long long)blockIdx.x * blockDim.x + threadIdx.x;
    if (t >= (long long)n * NHEAD) return;
    const int i = (int)(t >> 3), h = (int)(t & 7);
    const int sg = pairs ? pairs[2 * i] : seg[i];
    const float m = ord2f(mEnc[(size_t)sg * NHEAD + h]);
    const float e = expf(s[t] - m);
    s[t] = e;
    atomicAdd(&z[(size_t)sg * NHEAD + h], e);
}

// upd[seg,:] += (e/z[seg]) * V[b,:]
__global__ void scatter_wv(const float* __restrict__ e, const float* __restrict__ z,
                           const float* __restrict__ V, float* __restrict__ upd,
                           const int* __restrict__ pairs, const int* __restrict__ seg,
                           int n)
{
    const long long t = (long long)blockIdx.x * blockDim.x + threadIdx.x;
    if (t >= (long long)n * DMODEL) return;
    const int i = (int)(t >> 7), d = (int)(t & 127), h = d >> 4;
    int b2, sg;
    if (pairs) { sg = pairs[2 * i]; b2 = pairs[2 * i + 1]; }
    else       { sg = seg[i]; b2 = i; }
    const float w = e[(size_t)i * NHEAD + h] / z[(size_t)sg * NHEAD + h];
    atomicAdd(&upd[(size_t)sg * DMODEL + d], w * V[(size_t)b2 * DMODEL + d]);
}

// ---------------------------------------------------------------------------
extern "C" void kernel_launch(void* const* d_in, const int* in_sizes, int n_in,
                              void* d_out, int out_size, void* d_ws, size_t ws_size,
                              hipStream_t stream)
{
    (void)n_in; (void)out_size; (void)ws_size;
    const int NN = N_NODES, NE = N_EDGES;

    const float* node = (const float*)d_in[0];
    const float* edge = (const float*)d_in[1];
    const int*   eidx = (const int*)d_in[2];
    const int*   src  = eidx;
    const int*   dst  = eidx + NE;
    const int*   nei  = (const int*)d_in[3];
    const int    P    = in_sizes[3] / 2;

    // weights: index i -> d_in[4+2i] (_w), d_in[5+2i] (_b)
    // 0..6 : n_nq n_nk n_nv n_eq n_ek n_ev n_op   (128x128)
    // 7..13: e_nq e_nk e_nv e_eq e_ek e_ev e_op   (128x128)
    // 14 nf1(128x512) 15 nf2(512x128) 16 ef1(128x512) 17 ef2(512x128)
    static const int wdin[18]  = {128,128,128,128,128,128,128,
                                  128,128,128,128,128,128,128,
                                  128,512,128,512};
    static const int wdout[18] = {128,128,128,128,128,128,128,
                                  128,128,128,128,128,128,128,
                                  512,128,512,128};
    auto Wp = [&](int i) { return (const float*)d_in[4 + 2 * i]; };
    auto Bp = [&](int i) { return (const float*)d_in[5 + 2 * i]; };
    // LN params: 0 ln_n1_g,1 ln_n1_b,2 ln_e1_g,3 ln_e1_b,4 ln_nf_g,5 ln_nf_b,
    //            6 ln_n2_g,7 ln_n2_b,8 ln_e2_g,9 ln_e2_b,10 ln_ef_g,11 ln_ef_b
    auto LNp = [&](int j) { return (const float*)d_in[40 + j]; };

    // ---- workspace bump allocator (256B aligned) --------------------------
    char* base = (char*)d_ws;
    size_t off = 0;
    auto alloc = [&](size_t bytes) -> void* {
        void* p = base + off;
        off += (bytes + 255) & ~(size_t)255;
        return p;
    };

    _Float16* wt16[18];
    for (int i = 0; i < 18; ++i)
        wt16[i] = (_Float16*)alloc((size_t)wdin[i] * wdout[i] * sizeof(_Float16));

    _Float16* xn16 = (_Float16*)alloc((size_t)NN * DMODEL * 2);
    _Float16* xe16 = (_Float16*)alloc((size_t)NE * DMODEL * 2);
    float* nQ = (float*)alloc((size_t)NN * DMODEL * 4);
    float* nK = (float*)alloc((size_t)NN * DMODEL * 4);
    float* nV = (float*)alloc((size_t)NN * DMODEL * 4);
    float* Qe = (float*)alloc((size_t)NE * DMODEL * 4);   // reused block 2
    float* Ke = (float*)alloc((size_t)NE * DMODEL * 4);
    float* Ve = (float*)alloc((size_t)NE * DMODEL * 4);
    unsigned* mEnc_n = (unsigned*)alloc((size_t)NN * NHEAD * 4);
    float*    z_n    = (float*)alloc((size_t)NN * NHEAD * 4);
    float*    upd_n  = (float*)alloc((size_t)NN * DMODEL * 4);
    float*    s_e    = (float*)alloc((size_t)NE * NHEAD * 4);
    _Float16* upd16n = (_Float16*)alloc((size_t)NN * DMODEL * 2);
    float*    nprime = (float*)alloc((size_t)NN * DMODEL * 4);
    _Float16* npf16  = (_Float16*)alloc((size_t)NN * DMODEL * 2);
    _Float16* hn16   = (_Float16*)alloc((size_t)NN * HID * 2);
    unsigned* mEnc_e = (unsigned*)alloc((size_t)NE * NHEAD * 4);
    float*    z_e    = (float*)alloc((size_t)NE * NHEAD * 4);
    float*    upd_e  = (float*)alloc((size_t)NE * DMODEL * 4);
    float*    s_p    = (float*)alloc((size_t)P * NHEAD * 4);
    _Float16* upd16e = (_Float16*)alloc((size_t)NE * DMODEL * 2);
    float*    eprime = (float*)alloc((size_t)NE * DMODEL * 4);
    _Float16* epf16  = (_Float16*)alloc((size_t)NE * DMODEL * 2);
    _Float16* he16   = (_Float16*)alloc((size_t)NE * HID * 2);

    float* n_new = (float*)d_out;                       // [NN x 128]
    float* e_new = (float*)d_out + (size_t)NN * DMODEL; // [NE x 128]

    auto nb = [](long long n) { return (unsigned)((n + 255) / 256); };
    auto gemm = [&](const _Float16* A, int wi, const float* res, float* Cf,
                    _Float16* Ch, int M, int relu) {
        const int N = wdout[wi], K = wdin[wi];
        const long long waves = (long long)(M / 32) * (N / 32);  // 32x32 per wave
        gemm_wmma_f16<<<(unsigned)((waves + 7) / 8), 256, 0, stream>>>(
            A, wt16[wi], Bp(wi), res, Cf, Ch, M, N, K, relu);
    };

    // ---- weight convert+transpose ----------------------------------------
    for (int i = 0; i < 18; ++i) {
        const long long n = (long long)wdin[i] * wdout[i];
        w_to_f16t<<<nb(n), 256, 0, stream>>>(Wp(i), wt16[i], wdin[i], wdout[i]);
    }

    // ======================= block 1: node update ==========================
    ln_to_f16<<<nb((long long)NN * 32), 256, 0, stream>>>(node, LNp(0), LNp(1), xn16, NN);
    ln_to_f16<<<nb((long long)NE * 32), 256, 0, stream>>>(edge, LNp(2), LNp(3), xe16, NE);

    gemm(xn16, 0, nullptr, nQ, nullptr, NN, 0);
    gemm(xn16, 1, nullptr, nK, nullptr, NN, 0);
    gemm(xn16, 2, nullptr, nV, nullptr, NN, 0);
    gemm(xe16, 3, nullptr, Qe, nullptr, NE, 0);
    gemm(xe16, 4, nullptr, Ke, nullptr, NE, 0);
    gemm(xe16, 5, nullptr, Ve, nullptr, NE, 0);

    gather_add<<<nb((long long)NE * DMODEL), 256, 0, stream>>>(Qe, nQ, src, NE);
    gather_add<<<nb((long long)NE * DMODEL), 256, 0, stream>>>(Ke, nK, dst, NE);
    gather_add<<<nb((long long)NE * DMODEL), 256, 0, stream>>>(Ve, nV, dst, NE);

    hipMemsetAsync(mEnc_n, 0, (size_t)NN * NHEAD * 4, stream);
    hipMemsetAsync(z_n,    0, (size_t)NN * NHEAD * 4, stream);
    hipMemsetAsync(upd_n,  0, (size_t)NN * DMODEL * 4, stream);

    scores_k<<<nb((long long)NE * NHEAD), 256, 0, stream>>>(Qe, Ke, nullptr, src, s_e, mEnc_n, NE);
    expsum_k<<<nb((long long)NE * NHEAD), 256, 0, stream>>>(s_e, mEnc_n, z_n, nullptr, src, NE);
    scatter_wv<<<nb((long long)NE * DMODEL), 256, 0, stream>>>(s_e, z_n, Ve, upd_n, nullptr, src, NE);

    f32_to_f16<<<nb((long long)NN * DMODEL), 256, 0, stream>>>(upd_n, upd16n, NN * DMODEL);
    gemm(upd16n, 6, node, nprime, nullptr, NN, 0);                 // n_prime
    ln_to_f16<<<nb((long long)NN * 32), 256, 0, stream>>>(nprime, LNp(4), LNp(5), npf16, NN);
    gemm(npf16, 14, nullptr, nullptr, hn16, NN, 1);                // relu(nf1) -> f16
    gemm(hn16, 15, nprime, n_new, nullptr, NN, 0);                 // n_new -> d_out

    // ======================= block 2: edge update ==========================
    ln_to_f16<<<nb((long long)NN * 32), 256, 0, stream>>>(n_new, LNp(6), LNp(7), xn16, NN);
    ln_to_f16<<<nb((long long)NE * 32), 256, 0, stream>>>(edge,  LNp(8), LNp(9), xe16, NE);

    gemm(xn16, 7, nullptr, nQ, nullptr, NN, 0);
    gemm(xn16, 8, nullptr, nK, nullptr, NN, 0);
    gemm(xn16, 9, nullptr, nV, nullptr, NN, 0);
    gemm(xe16, 10, nullptr, Qe, nullptr, NE, 0);
    gemm(xe16, 11, nullptr, Ke, nullptr, NE, 0);
    gemm(xe16, 12, nullptr, Ve, nullptr, NE, 0);

    gather_add<<<nb((long long)NE * DMODEL), 256, 0, stream>>>(Qe, nQ, src, NE);
    gather_add<<<nb((long long)NE * DMODEL), 256, 0, stream>>>(Qe, nQ, dst, NE);
    gather_add<<<nb((long long)NE * DMODEL), 256, 0, stream>>>(Ke, nK, src, NE);
    gather_add<<<nb((long long)NE * DMODEL), 256, 0, stream>>>(Ke, nK, dst, NE);
    gather_add<<<nb((long long)NE * DMODEL), 256, 0, stream>>>(Ve, nV, src, NE);
    gather_add<<<nb((long long)NE * DMODEL), 256, 0, stream>>>(Ve, nV, dst, NE);

    hipMemsetAsync(mEnc_e, 0, (size_t)NE * NHEAD * 4, stream);
    hipMemsetAsync(z_e,    0, (size_t)NE * NHEAD * 4, stream);
    hipMemsetAsync(upd_e,  0, (size_t)NE * DMODEL * 4, stream);

    scores_k<<<nb((long long)P * NHEAD), 256, 0, stream>>>(Qe, Ke, nei, nullptr, s_p, mEnc_e, P);
    expsum_k<<<nb((long long)P * NHEAD), 256, 0, stream>>>(s_p, mEnc_e, z_e, nei, nullptr, P);
    scatter_wv<<<nb((long long)P * DMODEL), 256, 0, stream>>>(s_p, z_e, Ve, upd_e, nei, nullptr, P);

    f32_to_f16<<<nb((long long)NE * DMODEL), 256, 0, stream>>>(upd_e, upd16e, NE * DMODEL);
    gemm(upd16e, 13, edge, eprime, nullptr, NE, 0);                // e_prime
    ln_to_f16<<<nb((long long)NE * 32), 256, 0, stream>>>(eprime, LNp(10), LNp(11), epf16, NE);
    gemm(epf16, 16, nullptr, nullptr, he16, NE, 1);                // relu(ef1) -> f16
    gemm(he16, 17, eprime, e_new, nullptr, NE, 0);                 // e_new -> d_out
}
